// SelfAttention_74483322847633
// MI455X (gfx1250) — matrix operand order repaired and checked by reading, not compiled, not run on hardware
//
#include <hip/hip_runtime.h>

typedef __bf16 bf16_t;
typedef __attribute__((ext_vector_type(16))) __bf16 v16bf;
typedef __attribute__((ext_vector_type(8)))  __bf16 v8bf;
typedef __attribute__((ext_vector_type(8)))  float  v8f;
typedef __attribute__((ext_vector_type(4)))  unsigned v4u;
typedef __attribute__((ext_vector_type(8)))  int    v8i_t;
typedef __attribute__((ext_vector_type(4)))  int    v4i_t;

// Geometry
#define Bb   4
#define Ss   8
#define Cc   256
#define Aa   256
#define HW   4096
#define Pt   16            // spatial positions per workgroup
#define COLS 128           // Ss * Pt columns in the projection GEMM
#define STR  272           // bf16 elements per LDS row (256 + 16 pad, 32B-aligned rows)

// LDS layout (bytes)
#define PLANE_BYTES (COLS * STR * 2)            // 69632
#define X_OFF   0
#define Q_OFF   (PLANE_BYTES)                   // 69632
#define K_OFF   (2 * PLANE_BYTES)               // 139264 (also fp32 x staging)
#define V_OFF   (3 * PLANE_BYTES)               // 208896
#define BI_OFF  (4 * PLANE_BYTES)               // 278528  (3*256 floats)
#define WL_OFF  (BI_OFF + 3 * 256 * 4)          // 281600  (16*8*8 floats)
#define SMEM_BYTES (WL_OFF + Pt * Ss * Ss * 4)  // 285696

// ---------------------------------------------------------------------------
// Kernel 0: convert wq|wk|wv (fp32, row-major [A][C]) to bf16 in d_ws.
// ---------------------------------------------------------------------------
__global__ void __launch_bounds__(256)
wcvt_kernel(const float* __restrict__ wq, const float* __restrict__ wk,
            const float* __restrict__ wv, bf16_t* __restrict__ wb) {
    int i = blockIdx.x * 256 + threadIdx.x;          // 0 .. 3*65536-1
    const float* src = (i < 65536) ? wq : (i < 131072 ? wk : wv);
    wb[i] = (bf16_t)src[i & 65535];
}

// ---------------------------------------------------------------------------
// Kernel 1: TDM-staged x + fused QKV projection (bf16 WMMA) + attention.
// grid = (B * HW / Pt) = 1024 workgroups, 256 threads (8 wave32).
// ---------------------------------------------------------------------------
__global__ void __launch_bounds__(256)
attn_kernel(const float* __restrict__ x, const bf16_t* __restrict__ wb,
            const float* __restrict__ bq, const float* __restrict__ bk,
            const float* __restrict__ bv, float* __restrict__ out) {
    extern __shared__ __align__(32) char smem[];
    bf16_t* x_lds  = (bf16_t*)(smem + X_OFF);
    float*  bi_lds = (float*) (smem + BI_OFF);
    float*  w_lds  = (float*) (smem + WL_OFF);
    float*  xstage = (float*) (smem + K_OFF);    // 2048 x 16 fp32 (128 KB)

    const int tid   = threadIdx.x;
    const int b     = blockIdx.x >> 8;           // 4 batches
    const int pbase = (blockIdx.x & 255) << 4;   // 16 consecutive w-positions

    // ---- stage biases (fp32) ----
    bi_lds[tid]       = bq[tid];
    bi_lds[256 + tid] = bk[tid];
    bi_lds[512 + tid] = bv[tid];

    // ---- TDM: DMA the x tile (2048 rows of (s,c) x 16 fp32) into LDS ----
    if (tid < 32) {                              // wave 0 issues the tensor DMA
        unsigned long long gaddr =
            (unsigned long long)(const char*)(x + (size_t)(b * 2048) * HW + pbase);
        unsigned ldsa = (unsigned)(uintptr_t)(smem + K_OFF);

        v4u g0;                                  // D# group 0
        g0.x = 1u;                               // count=1, user descriptor
        g0.y = ldsa;                             // lds_addr (bytes)
        g0.z = (unsigned)(gaddr & 0xffffffffu);  // global_addr[31:0]
        g0.w = (unsigned)((gaddr >> 32) & 0x01ffffffu) | (2u << 30); // addr[56:32] | type=2

        v8i_t g1;                                // D# group 1
        g1[0] = (int)(2u << 16);                 // data_size=2 -> 4-byte elements
        g1[1] = (int)((4096u & 0xffffu) << 16);  // tensor_dim0[15:0]
        g1[2] = (int)((4096u >> 16) | ((2048u & 0xffffu) << 16)); // dim0 hi | tensor_dim1 lo
        g1[3] = (int)((2048u >> 16) | (16u << 16));               // dim1 hi | tile_dim0=16
        g1[4] = (int)(2048u);                    // tile_dim1=2048, tile_dim2=0
        g1[5] = (int)(4096u);                    // tensor_dim0_stride[31:0]
        g1[6] = 0;                               // stride hi | tensor_dim1_stride lo
        g1[7] = 0;

        v4i_t gz = {0, 0, 0, 0};                 // 2-D tensor: groups 2/3 disabled
#if __has_include(<hip/amd_detail/amd_gfx1250_TDM.h>)
        v8i_t gz8 = {0, 0, 0, 0, 0, 0, 0, 0};
        __builtin_amdgcn_tensor_load_to_lds(g0, g1, gz, gz, gz8, 0);
#else
        __builtin_amdgcn_tensor_load_to_lds(g0, g1, gz, gz, 0);
#endif
        __builtin_amdgcn_s_wait_tensorcnt(0);
    }
    __syncthreads();

    // ---- convert staged fp32 -> bf16 plane x_lds[col = s*16+p][c] ----
    {
        const int p  = tid & 15;
        const int rt = tid >> 4;                 // 16 (s,c)-rows per iteration
        for (int it = 0; it < 128; ++it) {
            int r = it * 16 + rt;                // r = s*256 + c
            float v = xstage[r * 16 + p];
            int s = r >> 8, c = r & 255;
            x_lds[(s * Pt + p) * STR + c] = (bf16_t)v;
        }
    }
    __syncthreads();

    // ---- QKV projection: [A=256 x C=256] (bf16) @ [C=256 x 128] -> LDS ----
    const int wave = tid >> 5;
    const int lane = tid & 31;
    const int lrow = lane & 15;
    const int sel  = lane >> 4;

    for (int job = wave; job < 48; job += 8) {   // 3 proj * 16 a-tiles
        const int proj = job >> 4;
        const int a0   = (job & 15) << 4;
        const bf16_t* wrow = wb + proj * 65536 + (a0 + lrow) * 256;

        // A fragments: 16x32 bf16 per k-step; lane<16 holds K{0..15},
        // lane>=16 holds K{16..31} of row M = lane&15.
        v16bf afr[8];
#pragma unroll
        for (int kk = 0; kk < 8; ++kk) {
            union { struct { v8bf lo, hi; } p; v16bf v; } u;
            u.p.lo = *(const v8bf*)(wrow + kk * 32 + sel * 16);
            u.p.hi = *(const v8bf*)(wrow + kk * 32 + sel * 16 + 8);
            afr[kk] = u.v;
        }

        bf16_t*      dst  = (bf16_t*)(smem + Q_OFF + proj * PLANE_BYTES);
        const float* bias = bi_lds + proj * 256;

#pragma unroll
        for (int ct = 0; ct < 8; ++ct) {         // 8 column tiles of 16 cols
            v8f acc = {};
            const bf16_t* bb = x_lds + (ct * 16 + lrow) * STR + sel * 16;
#pragma unroll
            for (int kk = 0; kk < 8; ++kk) {     // K = 256 in steps of 32
                v16bf bfr = *(const v16bf*)(bb + kk * 32);
                acc = __builtin_amdgcn_wmma_f32_16x16x32_bf16(
                    false, afr[kk], false, bfr, (short)0, acc, false, false);
            }
            // D layout: VGPR j -> row a0+j (lanes 0-15) / a0+8+j (lanes 16-31),
            // col = ct*16 + (lane&15). Store bf16-packed into proj plane [col][a].
            const int col = ct * 16 + lrow;
            const int ab  = a0 + sel * 8;
            unsigned* drow = (unsigned*)(dst + col * STR + ab);
#pragma unroll
            for (int j = 0; j < 8; j += 2) {
                union { bf16_t h[2]; unsigned u; } pk;
                pk.h[0] = (bf16_t)(acc[j]     + bias[ab + j]);
                pk.h[1] = (bf16_t)(acc[j + 1] + bias[ab + j + 1]);
                drow[j >> 1] = pk.u;
            }
        }
    }
    __syncthreads();

    // ---- attention scores + softmax (scrambled-index einsum, per (s,p)) ----
    // q_new[s][a] = Q[s'=a&7][32s + (a>>3)], k_new[a][t] = K[s'=a>>5][8(a&31)+t]
    if (tid < 128) {
        const int p = tid & 15, s = tid >> 4;
        const bf16_t* qld = (const bf16_t*)(smem + Q_OFF);
        const bf16_t* kld = (const bf16_t*)(smem + K_OFF);
        float acc[8] = {0.f, 0.f, 0.f, 0.f, 0.f, 0.f, 0.f, 0.f};
        for (int a = 0; a < 256; ++a) {
            const int u = a >> 5, w2 = a & 31;
            float qv = (float)qld[((w2 & 7) * Pt + p) * STR + 32 * s + 4 * u + (w2 >> 3)];
            const bf16_t* kr = kld + (u * Pt + p) * STR + 8 * w2;
#pragma unroll
            for (int t = 0; t < 8; ++t) acc[t] += qv * (float)kr[t];
        }
        const float scale = 0.0625f;             // 1/sqrt(256)
        float m = acc[0] * scale;
#pragma unroll
        for (int t = 1; t < 8; ++t) m = fmaxf(m, acc[t] * scale);
        float e[8], sum = 0.f;
#pragma unroll
        for (int t = 0; t < 8; ++t) { e[t] = __expf(acc[t] * scale - m); sum += e[t]; }
        const float inv = 1.0f / sum;
#pragma unroll
        for (int t = 0; t < 8; ++t) w_lds[(p * 8 + s) * 8 + t] = e[t] * inv;
    }
    __syncthreads();

    // ---- attended + output permute ----
    // out[b,so,ao,p] = sum_t w[p][ao>>5][t] * V[s'=so][32t + (ao&31)][p]
    {
        const int p  = tid & 15;
        const int ar = tid >> 4;                 // 16 a-rows per chunk
        const bf16_t* vld = (const bf16_t*)(smem + V_OFF);
        for (int so = 0; so < 8; ++so) {
            for (int ch = 0; ch < 16; ++ch) {
                const int ao = ch * 16 + ar;
                const float*  wv_ = w_lds + (p * 8 + (ao >> 5)) * 8;
                const bf16_t* vr  = vld + (so * Pt + p) * STR + (ao & 31);
                float acc = 0.f;
#pragma unroll
                for (int t = 0; t < 8; ++t) acc += wv_[t] * (float)vr[32 * t];
                out[((size_t)(b * 8 + so) * 256 + ao) * HW + pbase + p] = acc;
            }
        }
    }
}

// ---------------------------------------------------------------------------
extern "C" void kernel_launch(void* const* d_in, const int* in_sizes, int n_in,
                              void* d_out, int out_size, void* d_ws, size_t ws_size,
                              hipStream_t stream) {
    (void)in_sizes; (void)n_in; (void)out_size; (void)ws_size;
    const float* x  = (const float*)d_in[0];
    const float* wq = (const float*)d_in[1];
    const float* bq = (const float*)d_in[2];
    const float* wk = (const float*)d_in[3];
    const float* bk = (const float*)d_in[4];
    const float* wv = (const float*)d_in[5];
    const float* bv = (const float*)d_in[6];
    bf16_t* wb = (bf16_t*)d_ws;                  // 3*65536 bf16 = 384 KB

    wcvt_kernel<<<768, 256, 0, stream>>>(wq, wk, wv, wb);

    (void)hipFuncSetAttribute(reinterpret_cast<const void*>(attn_kernel),
                              hipFuncAttributeMaxDynamicSharedMemorySize,
                              SMEM_BYTES);
    attn_kernel<<<Bb * (HW / Pt), 256, SMEM_BYTES, stream>>>(
        x, wb, bq, bk, bv, (float*)d_out);
}
